// GATModel_63625645522948
// MI455X (gfx1250) — matrix operand order repaired and checked by reading, not compiled, not run on hardware
//
#include <hip/hip_runtime.h>
#include <cstdint>

typedef __attribute__((ext_vector_type(16))) _Float16 v16h;
typedef __attribute__((ext_vector_type(4)))  _Float16 v4h;
typedef __attribute__((ext_vector_type(8)))  float    v8f;

// ---------------------------------------------------------------------------
// helpers
// ---------------------------------------------------------------------------
__device__ __forceinline__ int f32_order(float f) {
  int i = __float_as_int(f);
  return (i >= 0) ? i : (i ^ 0x7fffffff);   // monotonic float->int map
}
__device__ __forceinline__ float f32_unorder(int i) {
  return __int_as_float((i >= 0) ? i : (i ^ 0x7fffffff));
}

__global__ void fill_i32_kernel(int* __restrict__ p, int v, int n) {
  int i = blockIdx.x * blockDim.x + threadIdx.x;
  if (i < n) p[i] = v;
}

// ---------------------------------------------------------------------------
// Pack weights B[K,N] (f32, row-major) into f16 WMMA B-fragment order:
//   Bp[nTile][kTile][lane][e],  lane = kg*16 + col, element e <-> K = kT*32+kg*16+e
// Weights are <=256KB packed -> L2-resident, loaded per-lane as one 32B read.
// ---------------------------------------------------------------------------
__global__ void pack_b_kernel(const float* __restrict__ B,
                              _Float16* __restrict__ Bp, int K, int N)
{
  int t = blockIdx.x * blockDim.x + threadIdx.x;
  if (t >= K * N) return;
  int k = t / N, n = t - k * N;
  int nT = n >> 4, col = n & 15;
  int kT = k >> 5, kk = k & 31;
  int kg = kk >> 4, e = kk & 15;
  int lane = kg * 16 + col;
  size_t idx = (((size_t)nT * (K >> 5) + kT) * 32 + lane) * 16 + e;
  Bp[idx] = (_Float16)B[t];
}

// ---------------------------------------------------------------------------
// WMMA GEMM: C[M,N] = A[M,K] @ B[K,N]
//   A: f32 row-major, staged to LDS in fragment order (f16)
//   B: pre-packed f16 fragment order (global, L2-resident)
// Block tile 128x64, BK=32. 256 threads = 8 waves; each wave owns a 32x32
// sub-tile -> 4 v_wmma_f32_16x16x32_f16 per K-step with fragment reuse.
// ---------------------------------------------------------------------------
#define GBM 128
#define GBN 64
#define GBK 32

__global__ __launch_bounds__(256) void gemm_wmma_f16(
    const float* __restrict__ A, const _Float16* __restrict__ Bp,
    float* __restrict__ C, int M, int N, int K)
{
  // A tile in per-lane fragment order: AsF[row][kg][e], 32B per (row,kg)
  __shared__ alignas(32) _Float16 AsF[GBM][2][16];   // 8 KB

  const int tid  = threadIdx.x;
  const int wave = tid >> 5;
  const int lane = tid & 31;
  const int mBlock = blockIdx.x * GBM;
  const int nBlock = blockIdx.y * GBN;
  const int wM = (wave & 3) << 5;    // 32-row band within block
  const int wN = (wave >> 2) << 5;   // 32-col band within block
  const int lrow = lane & 15;
  const int lkg  = lane >> 4;
  const int kTiles = K >> 5;

  v8f acc00 = {}, acc01 = {}, acc10 = {}, acc11 = {};

  for (int k0 = 0; k0 < K; k0 += GBK) {
    // ---- stage A tile: 128x32 f32 -> f16, branchless OOB (clamp + mask)
#pragma unroll
    for (int it = 0; it < 4; ++it) {
      int i  = tid + it * 256;         // float4 index: 128 rows x 8 quads
      int r  = i >> 3;
      int c4 = (i & 7) << 2;
      int gr = mBlock + r;
      int grc = gr < M ? gr : (M - 1);
      const float* ap = A + (size_t)grc * K + (k0 + c4);
      float x0 = ap[0], x1 = ap[1], x2 = ap[2], x3 = ap[3];
      float m = (gr < M) ? 1.f : 0.f;
      v4h h;
      h[0] = (_Float16)(x0 * m); h[1] = (_Float16)(x1 * m);
      h[2] = (_Float16)(x2 * m); h[3] = (_Float16)(x3 * m);
      int kg = c4 >> 4;
      int e  = c4 & 15;   // {0,4,8,12}
      *(v4h*)&AsF[r][kg][e] = h;
    }
    __syncthreads();

    // ---- fragments: contiguous 32B per lane
    v16h a0 = *(const v16h*)&AsF[wM + lrow][lkg][0];
    v16h a1 = *(const v16h*)&AsF[wM + 16 + lrow][lkg][0];
    const int kT = k0 >> 5;
    const int nT0 = (nBlock + wN) >> 4;
    v16h b0 = *(const v16h*)(Bp + (((size_t)nT0 * kTiles + kT) * 32 + lane) * 16);
    v16h b1 = *(const v16h*)(Bp + (((size_t)(nT0 + 1) * kTiles + kT) * 32 + lane) * 16);

    acc00 = __builtin_amdgcn_wmma_f32_16x16x32_f16(false, a0, false, b0, (short)0, acc00, false, false);
    acc01 = __builtin_amdgcn_wmma_f32_16x16x32_f16(false, a0, false, b1, (short)0, acc01, false, false);
    acc10 = __builtin_amdgcn_wmma_f32_16x16x32_f16(false, a1, false, b0, (short)0, acc10, false, false);
    acc11 = __builtin_amdgcn_wmma_f32_16x16x32_f16(false, a1, false, b1, (short)0, acc11, false, false);
    __syncthreads();
  }

  // ---- store D: lanes 0-15 -> M=v,N=lane ; lanes 16-31 -> M=v+8,N=lane-16
  // accN0 tiles write col, accN1 tiles write col+16.
  const int colBase = nBlock + wN + lrow;
  const int rowT    = mBlock + wM + (lkg << 3);
  v8f accL[2] = {acc00, acc10};   // left 16-col tile, mt = 0,1
  v8f accR[2] = {acc01, acc11};   // right 16-col tile

  if (mBlock + GBM <= M) {
    // fast path: whole block interior, unguarded stores, incremental addresses
#pragma unroll
    for (int mt = 0; mt < 2; ++mt) {
      float* cp = C + (size_t)(rowT + mt * 16) * N + colBase;
#pragma unroll
      for (int v = 0; v < 8; ++v) {
        cp[0]  = accL[mt][v];
        cp[16] = accR[mt][v];
        cp += N;
      }
    }
  } else {
#pragma unroll
    for (int mt = 0; mt < 2; ++mt) {
      float* cp = C + (size_t)(rowT + mt * 16) * N + colBase;
#pragma unroll
      for (int v = 0; v < 8; ++v) {
        if (rowT + mt * 16 + v < M) {
          cp[0]  = accL[mt][v];
          cp[16] = accR[mt][v];
        }
        cp += N;
      }
    }
  }
}

// ---------------------------------------------------------------------------
// self-loop attr: deg + segment_sum(edge_attr, dst); then divide.
// ---------------------------------------------------------------------------
__global__ void deg_loopsum_kernel(const int* __restrict__ dst,
    const float* __restrict__ ea, float* __restrict__ loopsum,
    float* __restrict__ deg, int E)
{
  long long t = (long long)blockIdx.x * blockDim.x + threadIdx.x;
  if (t >= (long long)E * 64) return;
  int e = (int)(t >> 6), k = (int)(t & 63);
  int d = dst[e];
  atomicAdd(&loopsum[(size_t)d * 64 + k], ea[(size_t)e * 64 + k]);
  if (k == 0) atomicAdd(&deg[d], 1.f);
}

__global__ void loop_attr_kernel(float* __restrict__ loopsum,
                                 const float* __restrict__ deg, int Nn)
{
  int t = blockIdx.x * blockDim.x + threadIdx.x;
  if (t >= Nn * 64) return;
  float dg = deg[t >> 6];
  dg = dg > 1.f ? dg : 1.f;
  loopsum[t] = loopsum[t] / dg;
}

// ---------------------------------------------------------------------------
// ve[k,h] = sum_c We[k, h*C+c] * ae[h,c]   (64 x 4, one block of 256)
// ---------------------------------------------------------------------------
__global__ void ve_kernel(const float* __restrict__ We,
                          const float* __restrict__ aeh,
                          float* __restrict__ ve, int HC, int C)
{
  int t = threadIdx.x;            // 0..255
  int k = t >> 2, h = t & 3;      // k<64, h<4
  float acc = 0.f;
  for (int c = 0; c < C; ++c)
    acc += We[(size_t)k * HC + h * C + c] * aeh[h * C + c];
  ve[k * 4 + h] = acc;
}

// ---------------------------------------------------------------------------
// per-node attention logits: al_s / al_d reductions over C
// ---------------------------------------------------------------------------
__global__ void node_al_kernel(const float* __restrict__ hs,
    const float* __restrict__ as_, const float* __restrict__ ad_,
    float* __restrict__ als, float* __restrict__ ald, int Nn, int HC, int C)
{
  int t = blockIdx.x * blockDim.x + threadIdx.x;
  if (t >= Nn * 4) return;
  int n = t >> 2, h = t & 3;
  const float* hp = hs + (size_t)n * HC + h * C;
  float a = 0.f, b = 0.f;
  for (int c = 0; c < C; ++c) {
    float v = hp[c];
    a += v * as_[h * C + c];
    b += v * ad_[h * C + c];
  }
  als[t] = a; ald[t] = b;
}

// ---------------------------------------------------------------------------
// per-edge alpha = leaky_relu(al_s[src]+al_d[dst]+ea.ve) ; atomicMax per dst
// ---------------------------------------------------------------------------
__global__ void edge_alpha_kernel(const int* __restrict__ src,
    const int* __restrict__ dst, const float* __restrict__ ea,
    const float* __restrict__ loop_attr, const float* __restrict__ ve,
    const float* __restrict__ als, const float* __restrict__ ald,
    float* __restrict__ alpha, int* __restrict__ amaxi, int E, int ET)
{
  int e = blockIdx.x * blockDim.x + threadIdx.x;
  if (e >= ET) return;
  int s, d; const float* eav;
  if (e < E) { s = src[e]; d = dst[e]; eav = ea + (size_t)e * 64; }
  else       { s = e - E;  d = s;      eav = loop_attr + (size_t)s * 64; }
  float a0 = 0.f, a1 = 0.f, a2 = 0.f, a3 = 0.f;
#pragma unroll 4
  for (int k = 0; k < 64; ++k) {
    float v = eav[k];
    a0 += v * ve[k * 4 + 0]; a1 += v * ve[k * 4 + 1];
    a2 += v * ve[k * 4 + 2]; a3 += v * ve[k * 4 + 3];
  }
  float ae4[4] = {a0, a1, a2, a3};
#pragma unroll
  for (int h = 0; h < 4; ++h) {
    float a = als[s * 4 + h] + ald[d * 4 + h] + ae4[h];
    a = (a > 0.f) ? a : 0.2f * a;
    alpha[(size_t)e * 4 + h] = a;
    atomicMax(&amaxi[d * 4 + h], f32_order(a));
  }
}

// ---------------------------------------------------------------------------
// per-edge weighted scatter: one wave (32 lanes) per edge.
// agg[dst] += hs[src]*exp(alpha - amax[dst]) ; denom[dst] += w
// ---------------------------------------------------------------------------
__global__ __launch_bounds__(256) void edge_agg_kernel(
    const int* __restrict__ src, const int* __restrict__ dst,
    const float* __restrict__ hs, const float* __restrict__ alpha,
    const int* __restrict__ amaxi, float* __restrict__ agg,
    float* __restrict__ denom, int E, int ET, int HC, int shC)
{
  int e = (blockIdx.x * blockDim.x + threadIdx.x) >> 5;
  int lane = threadIdx.x & 31;
  if (e >= ET) return;
  int s, d;
  if (e < E) { s = src[e]; d = dst[e]; } else { s = e - E; d = s; }
  float w[4];
#pragma unroll
  for (int h = 0; h < 4; ++h) {
    float am = f32_unorder(amaxi[d * 4 + h]);
    w[h] = __expf(alpha[(size_t)e * 4 + h] - am);
  }
  if (lane < 4) atomicAdd(&denom[d * 4 + lane], w[lane]);
  for (int c = lane; c < HC; c += 32) {
    int h = c >> shC;
    atomicAdd(&agg[(size_t)d * HC + c], hs[(size_t)s * HC + c] * w[h]);
  }
}

// ---------------------------------------------------------------------------
// finalize: mode 0 -> concat + bias (+optional relu) [N,HC]
//           mode 1 -> mean over 4 heads + bias       [N,C]
// ---------------------------------------------------------------------------
__global__ void node_finalize_kernel(const float* __restrict__ agg,
    const float* __restrict__ denom, const float* __restrict__ bias,
    float* __restrict__ out, int Nn, int HC, int shHC, int shC,
    int mode, int relu)
{
  int t = blockIdx.x * blockDim.x + threadIdx.x;
  if (mode == 0) {
    if (t >= (Nn << shHC)) return;
    int n = t >> shHC, c = t & (HC - 1), h = c >> shC;
    float v = agg[t] / (denom[n * 4 + h] + 1e-16f) + bias[c];
    out[t] = (relu && v < 0.f) ? 0.f : v;
  } else {
    if (t >= (Nn << shC)) return;
    int C = 1 << shC;
    int n = t >> shC, c = t & (C - 1);
    float a = 0.f;
#pragma unroll
    for (int h = 0; h < 4; ++h)
      a += agg[((size_t)n << shHC) + h * C + c] / (denom[n * 4 + h] + 1e-16f);
    out[t] = a * 0.25f + bias[c];
  }
}

// ---------------------------------------------------------------------------
// batch mean pool + fused heads
// ---------------------------------------------------------------------------
__global__ void pool_kernel(const float* __restrict__ h3,
    const int* __restrict__ batch, float* __restrict__ pool,
    float* __restrict__ cnt, int Nn)
{
  int t = blockIdx.x * blockDim.x + threadIdx.x;
  if (t >= Nn * 128) return;
  int n = t >> 7, c = t & 127;
  int b = batch[n];
  atomicAdd(&pool[b * 128 + c], h3[t]);
  if (c == 0) atomicAdd(&cnt[b], 1.f);
}

__global__ void head_kernel(const float* __restrict__ pool,
    const float* __restrict__ cnt, const float* __restrict__ Wa,
    const float* __restrict__ ba, const float* __restrict__ Wv,
    const float* __restrict__ bv, float* __restrict__ out, int B)
{
  int t = blockIdx.x * blockDim.x + threadIdx.x;
  if (t >= B * 257) return;
  int b = t / 257, o = t - b * 257;
  float cn = cnt[b]; cn = cn > 1.f ? cn : 1.f;
  float acc = 0.f;
  for (int c = 0; c < 128; ++c) {
    float g = pool[b * 128 + c] / cn;
    acc += g * (o < 256 ? Wa[c * 256 + o] : Wv[c]);
  }
  out[t] = acc + (o < 256 ? ba[o] : bv[0]);
}

// ---------------------------------------------------------------------------
// launcher
// ---------------------------------------------------------------------------
static inline int ceil_div(long long a, long long b) { return (int)((a + b - 1) / b); }

extern "C" void kernel_launch(void* const* d_in, const int* in_sizes, int n_in,
                              void* d_out, int out_size, void* d_ws, size_t ws_size,
                              hipStream_t stream)
{
  (void)n_in; (void)out_size; (void)ws_size;
  const float* x     = (const float*)d_in[0];
  const int*   eidx  = (const int*)d_in[1];
  const float* ea    = (const float*)d_in[2];
  const int*   batch = (const int*)d_in[3];
  const float* W1  = (const float*)d_in[4];
  const float* as1 = (const float*)d_in[5];
  const float* ad1 = (const float*)d_in[6];
  const float* We1 = (const float*)d_in[7];
  const float* ae1 = (const float*)d_in[8];
  const float* b1  = (const float*)d_in[9];
  const float* W2  = (const float*)d_in[10];
  const float* as2 = (const float*)d_in[11];
  const float* ad2 = (const float*)d_in[12];
  const float* We2 = (const float*)d_in[13];
  const float* ae2 = (const float*)d_in[14];
  const float* b2  = (const float*)d_in[15];
  const float* W3  = (const float*)d_in[16];
  const float* as3 = (const float*)d_in[17];
  const float* ad3 = (const float*)d_in[18];
  const float* We3 = (const float*)d_in[19];
  const float* ae3 = (const float*)d_in[20];
  const float* b3  = (const float*)d_in[21];
  const float* Wa  = (const float*)d_in[22];
  const float* ba  = (const float*)d_in[23];
  const float* Wv  = (const float*)d_in[24];
  const float* bv  = (const float*)d_in[25];

  const int Nn = in_sizes[3];
  const int E  = in_sizes[1] / 2;
  const int ET = E + Nn;
  const int B  = 64;
  const int D_IN = in_sizes[0] / Nn;   // 192

  const int* src = eidx;
  const int* dst = eidx + E;

  // workspace layout
  float* p = (float*)d_ws;
  float* hs    = p; p += (size_t)Nn * 512;
  float* agg   = p; p += (size_t)Nn * 512;
  float* hA    = p; p += (size_t)Nn * 256;
  float* hB    = p; p += (size_t)Nn * 256;
  float* loopa = p; p += (size_t)Nn * 64;
  float* deg   = p; p += Nn;
  float* als   = p; p += (size_t)Nn * 4;
  float* ald   = p; p += (size_t)Nn * 4;
  float* denom = p; p += (size_t)Nn * 4;
  int*   amaxi = (int*)p; p += (size_t)Nn * 4;
  float* alpha = p; p += (size_t)ET * 4;
  float* ve    = p; p += 256;
  float* pool  = p; p += (size_t)B * 128;
  float* cnt   = p; p += B;
  _Float16* Bp = (_Float16*)p;         // packed weights, max 256*512 f16

  // self-loop attrs (PyG fill_value='mean')
  hipMemsetAsync(loopa, 0, (size_t)Nn * 64 * sizeof(float), stream);
  hipMemsetAsync(deg,   0, (size_t)Nn * sizeof(float), stream);
  deg_loopsum_kernel<<<ceil_div((long long)E * 64, 256), 256, 0, stream>>>(
      dst, ea, loopa, deg, E);
  loop_attr_kernel<<<ceil_div((long long)Nn * 64, 256), 256, 0, stream>>>(
      loopa, deg, Nn);

  auto log2i = [](int v) { int s = 0; while ((1 << s) < v) ++s; return s; };

  auto run_layer = [&](const float* hin, int K, int HC, int C,
                       const float* W, const float* as_, const float* ad_,
                       const float* We, const float* aeh, const float* bias,
                       float* hout, int mode, int relu)
  {
    const int shC = log2i(C), shHC = log2i(HC);
    pack_b_kernel<<<ceil_div((long long)K * HC, 256), 256, 0, stream>>>(W, Bp, K, HC);
    dim3 gg(ceil_div(Nn, GBM), HC / GBN);
    gemm_wmma_f16<<<gg, 256, 0, stream>>>(hin, Bp, hs, Nn, HC, K);
    ve_kernel<<<1, 256, 0, stream>>>(We, aeh, ve, HC, C);
    node_al_kernel<<<ceil_div((long long)Nn * 4, 256), 256, 0, stream>>>(
        hs, as_, ad_, als, ald, Nn, HC, C);
    fill_i32_kernel<<<ceil_div((long long)Nn * 4, 256), 256, 0, stream>>>(
        amaxi, (int)0x80000000, Nn * 4);
    hipMemsetAsync(denom, 0, (size_t)Nn * 4 * sizeof(float), stream);
    hipMemsetAsync(agg,   0, (size_t)Nn * HC * sizeof(float), stream);
    edge_alpha_kernel<<<ceil_div(ET, 256), 256, 0, stream>>>(
        src, dst, ea, loopa, ve, als, ald, alpha, amaxi, E, ET);
    edge_agg_kernel<<<ceil_div((long long)ET * 32, 256), 256, 0, stream>>>(
        src, dst, hs, alpha, amaxi, agg, denom, E, ET, HC, shC);
    int outElems = (mode == 0) ? (Nn << shHC) : (Nn << shC);
    node_finalize_kernel<<<ceil_div(outElems, 256), 256, 0, stream>>>(
        agg, denom, bias, hout, Nn, HC, shHC, shC, mode, relu);
  };

  run_layer(x,  D_IN, 256, 64,  W1, as1, ad1, We1, ae1, b1, hA, 0, 1);
  run_layer(hA, 256,  256, 64,  W2, as2, ad2, We2, ae2, b2, hB, 0, 1);
  run_layer(hB, 256,  512, 128, W3, as3, ad3, We3, ae3, b3, hA, 1, 0); // hA := h3 [N,128]

  hipMemsetAsync(pool, 0, (size_t)B * 128 * sizeof(float), stream);
  hipMemsetAsync(cnt,  0, (size_t)B * sizeof(float), stream);
  pool_kernel<<<ceil_div((long long)Nn * 128, 256), 256, 0, stream>>>(
      hA, batch, pool, cnt, Nn);
  head_kernel<<<ceil_div(B * 257, 256), 256, 0, stream>>>(
      pool, cnt, Wa, ba, Wv, bv, (float*)d_out, B);
}
